// DenseRegDSSResNet_52493090291992
// MI455X (gfx1250) — compile-verified, matrix operand order checked by reading, not run on hardware
//
#include <hip/hip_runtime.h>

#define BB 32
#define LL 2048
#define DIN 5
#define DOUT 5
#define HH 256
#define NN 32
#define NLAYERS 4
#define NRES 3

typedef __bf16 bf16;
typedef __attribute__((ext_vector_type(16))) __bf16 v16bf;
typedef __attribute__((ext_vector_type(8)))  float  v8f;

static __device__ __forceinline__ v8f wmma_bf(v16bf a, v16bf b, v8f c) {
  // D = A(16x32 bf16) * B(32x16 bf16) + C(16x16 f32)
  return __builtin_amdgcn_wmma_f32_16x16x32_bf16(false, a, false, b, (short)0, c,
                                                 false, false);
}

// ---------------------------------------------------------------------------
// Weight packing: conv3 weights (layer,O,I,T) f32 -> (layer,O,K) bf16, K=t*256+i
// ---------------------------------------------------------------------------
__global__ void pack_conv3_kernel(const float* __restrict__ src,
                                  bf16* __restrict__ dst, int total) {
  int idx = blockIdx.x * 256 + threadIdx.x;
  if (idx >= total) return;
  int k = idx % 768;
  int rest = idx / 768;
  int o = rest % HH;
  int layer = rest / HH;
  int t = k >> 8, i = k & 255;
  dst[idx] = (bf16)src[(((long)layer * HH + o) * HH + i) * 3 + t];
}

__global__ void pack_glu_kernel(const float* __restrict__ src,
                                bf16* __restrict__ dst, int total) {
  int idx = blockIdx.x * 256 + threadIdx.x;
  if (idx >= total) return;
  dst[idx] = (bf16)src[idx];  // already (layer, 2H, H) row-major
}

// ---------------------------------------------------------------------------
// Encoder conv: (B,5,L) k=15 pad7 -> h f32 + bf16 copy
// ---------------------------------------------------------------------------
__global__ __launch_bounds__(256) void enc_conv_kernel(
    const float* __restrict__ x, const float* __restrict__ w,
    const float* __restrict__ bias, float* __restrict__ hbuf,
    bf16* __restrict__ abf) {
  __shared__ float xl[DIN][78];  // 64 + 14
  const int b = blockIdx.y, l0 = blockIdx.x * 64, tid = threadIdx.x;
  for (int idx = tid; idx < DIN * 78; idx += 256) {
    int i = idx / 78, c = idx % 78;
    int l = l0 - 7 + c;
    xl[i][c] = (l >= 0 && l < LL) ? x[((long)b * DIN + i) * LL + l] : 0.f;
  }
  __syncthreads();
  const int o = tid;
  float wv[DIN * 15];
#pragma unroll
  for (int k = 0; k < DIN * 15; ++k) wv[k] = w[o * DIN * 15 + k];
  const float bia = bias[o];
  for (int p = 0; p < 64; ++p) {
    float acc = bia;
#pragma unroll
    for (int i = 0; i < DIN; ++i)
#pragma unroll
      for (int t = 0; t < 15; ++t) acc += wv[i * 15 + t] * xl[i][p + t];
    long idx = ((long)b * HH + o) * LL + l0 + p;
    hbuf[idx] = acc;
    abf[idx] = (bf16)acc;
  }
}

// ---------------------------------------------------------------------------
// WMMA conv k=3 pad1, 256->256 channels, always ReLU on output.
// K = 3*256 tap-major. 256 threads = 8 waves. Block tile: 256 oc x 32 pos.
// Each wave: 2 oc-tiles x 2 pos-tiles (A fragments reused across pos-tiles).
// LDS holds the input tile TRANSPOSED ([pos+halo][channel]) so B fragments
// are contiguous 32B -> ds_load_b128 instead of 16 scalar u16 gathers.
// ---------------------------------------------------------------------------
template <bool HAS_RES, bool HAS_F, bool HAS_B>
__global__ __launch_bounds__(256) void conv3_wmma_kernel(
    const bf16* __restrict__ in_bf, const bf16* __restrict__ wp,
    const float* __restrict__ bias, const float* __restrict__ res,
    float* __restrict__ out_f, bf16* __restrict__ out_b) {
  __shared__ __align__(32) bf16 ilds[34 * 256];  // [pos 0..33][channel]
  const int b = blockIdx.y;
  const int l0 = blockIdx.x * 32;
  const int tid = threadIdx.x;
  {
    const bf16* src = in_bf + ((long)b * HH + tid) * LL;
#pragma unroll
    for (int c = 0; c < 34; ++c) {
      int l = l0 - 1 + c;
      ilds[c * 256 + tid] = (l >= 0 && l < LL) ? src[l] : (bf16)0.0f;
    }
  }
  __syncthreads();
  const int lane = tid & 31;
  const int wid = tid >> 5;
  const int m = lane & 15;
  const int half = lane >> 4;
  const int n = lane & 15;
  const int oc0 = wid * 32;
  v8f acc00 = {}, acc01 = {}, acc10 = {}, acc11 = {};
  for (int kb = 0; kb < 768; kb += 32) {
    const int t = kb >> 8;
    const int i0 = (kb & 255) + half * 16;
    v16bf b0 = *(const v16bf*)&ilds[(n + t) * 256 + i0];
    v16bf b1 = *(const v16bf*)&ilds[(n + 16 + t) * 256 + i0];
    const bf16* arow0 = wp + (long)(oc0 + m) * 768 + kb + half * 16;
    __builtin_prefetch(arow0 + 32, 0, 1);
    v16bf a0 = *(const v16bf*)arow0;
    v16bf a1 = *(const v16bf*)(arow0 + 16 * 768);
    acc00 = wmma_bf(a0, b0, acc00);
    acc01 = wmma_bf(a0, b1, acc01);
    acc10 = wmma_bf(a1, b0, acc10);
    acc11 = wmma_bf(a1, b1, acc11);
  }
#pragma unroll
  for (int r = 0; r < 8; ++r) {
    const int mm = r + half * 8;  // C/D layout: lanes>=16 hold M = r+8
    const int o0 = oc0 + mm;
    const int o1 = oc0 + 16 + mm;
    const int la = l0 + n;
    const int lb = l0 + 16 + n;
    const long g00 = ((long)b * HH + o0) * LL + la;
    const long g01 = ((long)b * HH + o0) * LL + lb;
    const long g10 = ((long)b * HH + o1) * LL + la;
    const long g11 = ((long)b * HH + o1) * LL + lb;
    float v00 = acc00[r] + bias[o0];
    float v01 = acc01[r] + bias[o0];
    float v10 = acc10[r] + bias[o1];
    float v11 = acc11[r] + bias[o1];
    if (HAS_RES) {
      v00 += res[g00]; v01 += res[g01]; v10 += res[g10]; v11 += res[g11];
    }
    v00 = fmaxf(v00, 0.f); v01 = fmaxf(v01, 0.f);
    v10 = fmaxf(v10, 0.f); v11 = fmaxf(v11, 0.f);
    if (HAS_F) {
      out_f[g00] = v00; out_f[g01] = v01; out_f[g10] = v10; out_f[g11] = v11;
    }
    if (HAS_B) {
      out_b[g00] = (bf16)v00; out_b[g01] = (bf16)v01;
      out_b[g10] = (bf16)v10; out_b[g11] = (bf16)v11;
    }
  }
}

// ---------------------------------------------------------------------------
// DSS bidirectional diagonal recurrence (exact equivalent of the FFT conv):
//   y[l] = Re(sum_n W0_n s_n[l]) + Re(sum_n W1_n t_n[l]) + D*x[l]
//   s_n[l] = r_n s_n[l-1] + x[l],  t_n[l] = x[l+1] + r_n t_n[l+1],
//   r_n = exp(dt*lam_n).  One wave per (b,h) channel, lane = state n.
// Writes gelu(y) as bf16 for the GLU GEMM.
// ---------------------------------------------------------------------------
__global__ __launch_bounds__(256) void dss_scan_kernel(
    const float* __restrict__ hbuf, float* __restrict__ yfwd,
    bf16* __restrict__ ygelu, const float* __restrict__ log_dt,
    const float* __restrict__ lam_re, const float* __restrict__ lam_im,
    const float* __restrict__ W_re, const float* __restrict__ W_im,
    const float* __restrict__ Dp) {
  const int lane = threadIdx.x & 31;
  const int gw = (blockIdx.x * 256 + threadIdx.x) >> 5;
  if (gw >= BB * HH) return;
  const int b = gw / HH, hh = gw % HH;
  const float dtv = __expf(log_dt[hh]);
  const float ar = dtv * lam_re[lane], ai = dtv * lam_im[lane];
  const float mag = __expf(ar);
  const float rre = mag * __cosf(ai), rim = mag * __sinf(ai);
  const float w0r = W_re[(0 * HH + hh) * NN + lane];
  const float w0i = W_im[(0 * HH + hh) * NN + lane];
  const float w1r = W_re[(1 * HH + hh) * NN + lane];
  const float w1i = W_im[(1 * HH + hh) * NN + lane];
  const float Dv = Dp[hh];
  const float* xrow = hbuf + ((long)b * HH + hh) * LL;
  float* yf = yfwd + ((long)b * HH + hh) * LL;
  bf16* yo = ygelu + ((long)b * HH + hh) * LL;

  float sre = 0.f, sim = 0.f;
  for (int c = 0; c < LL; c += 32) {
    const float xv = xrow[c + lane];
    float yout = 0.f;
#pragma unroll
    for (int j = 0; j < 32; ++j) {
      float xj = __shfl(xv, j, 32);
      float nre = rre * sre - rim * sim + xj;
      sim = rre * sim + rim * sre;
      sre = nre;
      float yv = w0r * sre - w0i * sim;
#pragma unroll
      for (int off = 16; off; off >>= 1) yv += __shfl_xor(yv, off, 32);
      if (lane == j) yout = yv;
    }
    yf[c + lane] = yout;
  }

  float tre = 0.f, tim = 0.f, carry = 0.f;
  for (int c = LL - 32; c >= 0; c -= 32) {
    const float xv = xrow[c + lane];
    float yout = 0.f;
#pragma unroll
    for (int j = 31; j >= 0; --j) {
      float xn = (j == 31) ? carry : __shfl(xv, j + 1, 32);
      float nre = rre * tre - rim * tim + xn;
      tim = rre * tim + rim * tre;
      tre = nre;
      float yv = w1r * tre - w1i * tim;
#pragma unroll
      for (int off = 16; off; off >>= 1) yv += __shfl_xor(yv, off, 32);
      if (lane == j) yout = yv;
    }
    carry = __shfl(xv, 0, 32);
    float yt = yf[c + lane] + yout + Dv * xv;
    float yg = 0.5f * yt * (1.f + erff(yt * 0.70710678118f));  // exact gelu
    yo[c + lane] = (bf16)yg;
  }
}

// ---------------------------------------------------------------------------
// Fused GLU (512x256 WMMA GEMM) + sigmoid gate + residual + channel LayerNorm.
// 512 threads = 16 waves; 16 positions per block; K = 256.
// y tile staged transposed ([pos][channel]) -> contiguous v16bf B loads.
// ---------------------------------------------------------------------------
__global__ __launch_bounds__(512) void glu_ln_kernel(
    const bf16* __restrict__ yg, const bf16* __restrict__ wp,
    const float* __restrict__ gb, const float* __restrict__ lng,
    const float* __restrict__ lnb, float* __restrict__ hbuf) {
  __shared__ __align__(32) bf16 ylds[16 * HH];  // [pos][channel]
  __shared__ float zlds[2 * HH * 16];
  const int b = blockIdx.y, l0 = blockIdx.x * 16, tid = threadIdx.x;
  for (int idx = tid; idx < HH * 16; idx += 512) {
    int ch = idx >> 4, col = idx & 15;  // coalesced global read, LDS scatter
    ylds[col * HH + ch] = yg[((long)b * HH + ch) * LL + l0 + col];
  }
  __syncthreads();
  const int lane = tid & 31;
  const int wid = tid >> 5;
  const int m = lane & 15;
  const int half = lane >> 4;
  const int n = lane & 15;
  const int oc0 = wid * 32;
  const int oc1 = oc0 + 16;
  v8f acc0 = {};
  v8f acc1 = {};
  for (int kb = 0; kb < 256; kb += 32) {
    v16bf bv = *(const v16bf*)&ylds[n * HH + kb + half * 16];
    const bf16* arow0 = wp + (long)(oc0 + m) * 256 + kb + half * 16;
    __builtin_prefetch(arow0 + 32, 0, 1);
    v16bf a0 = *(const v16bf*)arow0;
    v16bf a1 = *(const v16bf*)(arow0 + 16 * 256);
    acc0 = wmma_bf(a0, bv, acc0);
    acc1 = wmma_bf(a1, bv, acc1);
  }
#pragma unroll
  for (int r = 0; r < 8; ++r) {
    int mm = r + half * 8;
    zlds[(oc0 + mm) * 16 + n] = acc0[r] + gb[oc0 + mm];
    zlds[(oc1 + mm) * 16 + n] = acc1[r] + gb[oc1 + mm];
  }
  __syncthreads();
  // phase 2: gate + residual + LayerNorm over 256 channels, per position.
  const int pn = tid >> 5;  // position 0..15 (constant within wave)
  const int cl = tid & 31;
  float u[8], s1 = 0.f, s2 = 0.f;
#pragma unroll
  for (int q = 0; q < 8; ++q) {
    int c = cl + 32 * q;
    float top = zlds[c * 16 + pn];
    float gate = zlds[(c + 256) * 16 + pn];
    float g = top / (1.f + __expf(-gate));
    float uu = g + hbuf[((long)b * HH + c) * LL + l0 + pn];
    u[q] = uu;
    s1 += uu;
    s2 += uu * uu;
  }
#pragma unroll
  for (int off = 16; off; off >>= 1) {
    s1 += __shfl_xor(s1, off, 32);
    s2 += __shfl_xor(s2, off, 32);
  }
  const float mu = s1 * (1.f / 256.f);
  const float var = s2 * (1.f / 256.f) - mu * mu;
  const float rstd = rsqrtf(var + 1e-5f);
#pragma unroll
  for (int q = 0; q < 8; ++q) {
    int c = cl + 32 * q;
    hbuf[((long)b * HH + c) * LL + l0 + pn] =
        (u[q] - mu) * rstd * lng[c] + lnb[c];
  }
}

// ---------------------------------------------------------------------------
// Species embedding add (in place), seq_emb output, mean-pool to agg.
// ---------------------------------------------------------------------------
__global__ __launch_bounds__(256) void species_agg_kernel(
    float* __restrict__ hbuf, const int* __restrict__ xs,
    const float* __restrict__ spe, float* __restrict__ seq_out,
    float* __restrict__ agg) {
  const int lane = threadIdx.x & 31;
  const int gw = (blockIdx.x * 256 + threadIdx.x) >> 5;
  if (gw >= BB * HH) return;
  const int b = gw / HH, ch = gw % HH;
  const float emb = spe[xs[b] * HH + ch];
  float* row = hbuf + ((long)b * HH + ch) * LL;
  float* srow = seq_out + ((long)b * HH + ch) * LL;
  float s = 0.f;
  for (int c = lane; c < LL; c += 32) {
    float v = row[c] + emb;
    row[c] = v;
    srow[c] = v;
    s += v;
  }
#pragma unroll
  for (int off = 16; off; off >>= 1) s += __shfl_xor(s, off, 32);
  if (lane == 0) agg[b * HH + ch] = s * (1.f / (float)LL);
}

// ---------------------------------------------------------------------------
// Tiny ELU MLP head: 256->128->64->32->1 for 32 rows. One block.
// ---------------------------------------------------------------------------
__global__ __launch_bounds__(256) void mlp_kernel(
    const float* __restrict__ agg, const float* __restrict__ w1,
    const float* __restrict__ b1, const float* __restrict__ w2,
    const float* __restrict__ b2, const float* __restrict__ w3,
    const float* __restrict__ b3, const float* __restrict__ w4,
    const float* __restrict__ b4, float* __restrict__ reg_out) {
  __shared__ float a[32 * 256];
  __shared__ float r1[32 * 128];
  __shared__ float r2[32 * 64];
  __shared__ float r3[32 * 32];
  const int tid = threadIdx.x;
  for (int i = tid; i < 32 * 256; i += 256) a[i] = agg[i];
  __syncthreads();
  for (int i = tid; i < 32 * 128; i += 256) {
    int bb = i >> 7, j = i & 127;
    float s = b1[j];
    for (int k = 0; k < 256; ++k) s += a[bb * 256 + k] * w1[j * 256 + k];
    r1[i] = s > 0.f ? s : __expf(s) - 1.f;
  }
  __syncthreads();
  for (int i = tid; i < 32 * 64; i += 256) {
    int bb = i >> 6, j = i & 63;
    float s = b2[j];
    for (int k = 0; k < 128; ++k) s += r1[bb * 128 + k] * w2[j * 128 + k];
    r2[i] = s > 0.f ? s : __expf(s) - 1.f;
  }
  __syncthreads();
  for (int i = tid; i < 32 * 32; i += 256) {
    int bb = i >> 5, j = i & 31;
    float s = b3[j];
    for (int k = 0; k < 64; ++k) s += r2[bb * 64 + k] * w3[j * 64 + k];
    r3[i] = s > 0.f ? s : __expf(s) - 1.f;
  }
  __syncthreads();
  if (tid < 32) {
    float s = b4[0];
    for (int k = 0; k < 32; ++k) s += r3[tid * 32 + k] * w4[k];
    reg_out[tid] = s;
  }
}

// ---------------------------------------------------------------------------
// Decoder conv: 256 -> 5, k=15, pad 7. Channel-chunked LDS staging.
// ---------------------------------------------------------------------------
__global__ __launch_bounds__(256) void dec_conv_kernel(
    const float* __restrict__ hbuf, const float* __restrict__ dw,
    const float* __restrict__ db, float* __restrict__ dec_out) {
  __shared__ float xl[32 * 270];  // 256 pos + 14 halo
  __shared__ float wl[DOUT * 32 * 15];
  const int b = blockIdx.y, l0 = blockIdx.x * 256, p = threadIdx.x;
  float acc[DOUT];
#pragma unroll
  for (int o = 0; o < DOUT; ++o) acc[o] = db[o];
  for (int cb = 0; cb < HH; cb += 32) {
    __syncthreads();
    for (int idx = p; idx < 32 * 270; idx += 256) {
      int i = idx / 270, c = idx % 270;
      int l = l0 - 7 + c;
      xl[idx] = (l >= 0 && l < LL) ? hbuf[((long)b * HH + cb + i) * LL + l] : 0.f;
    }
    for (int idx = p; idx < DOUT * 32 * 15; idx += 256) {
      int o = idx / (32 * 15);
      int rem = idx % (32 * 15);
      int i = rem / 15, t = rem % 15;
      wl[idx] = dw[((long)o * HH + cb + i) * 15 + t];
    }
    __syncthreads();
    for (int i = 0; i < 32; ++i) {
#pragma unroll
      for (int t = 0; t < 15; ++t) {
        float v = xl[i * 270 + p + t];
#pragma unroll
        for (int o = 0; o < DOUT; ++o) acc[o] += wl[(o * 32 + i) * 15 + t] * v;
      }
    }
  }
#pragma unroll
  for (int o = 0; o < DOUT; ++o)
    dec_out[((long)b * DOUT + o) * LL + l0 + p] = acc[o];
}

// ---------------------------------------------------------------------------
extern "C" void kernel_launch(void* const* d_in, const int* in_sizes, int n_in,
                              void* d_out, int out_size, void* d_ws,
                              size_t ws_size, hipStream_t stream) {
  (void)in_sizes; (void)n_in; (void)out_size; (void)ws_size;
  const float* x      = (const float*)d_in[0];
  const int*   xs     = (const int*)d_in[1];
  const float* enc_w  = (const float*)d_in[2];
  const float* enc_b  = (const float*)d_in[3];
  const float* res_w1 = (const float*)d_in[4];
  const float* res_b1 = (const float*)d_in[5];
  const float* res_w2 = (const float*)d_in[6];
  const float* res_b2 = (const float*)d_in[7];
  const float* log_dt = (const float*)d_in[8];
  const float* lam_re = (const float*)d_in[9];
  const float* lam_im = (const float*)d_in[10];
  const float* W_re   = (const float*)d_in[11];
  const float* W_im   = (const float*)d_in[12];
  const float* Dp     = (const float*)d_in[13];
  const float* glu_w  = (const float*)d_in[14];
  const float* glu_b  = (const float*)d_in[15];
  const float* ln_g   = (const float*)d_in[16];
  const float* ln_b   = (const float*)d_in[17];
  const float* dec_w  = (const float*)d_in[18];
  const float* dec_b  = (const float*)d_in[19];
  const float* sp_emb = (const float*)d_in[20];
  const float* h1w = (const float*)d_in[21];
  const float* h1b = (const float*)d_in[22];
  const float* h2w = (const float*)d_in[23];
  const float* h2b = (const float*)d_in[24];
  const float* h3w = (const float*)d_in[25];
  const float* h3b = (const float*)d_in[26];
  const float* h4w = (const float*)d_in[27];
  const float* h4b = (const float*)d_in[28];

  // Workspace layout (bytes):
  //   [0,64M)     hbuf f32 activations
  //   [64M,128M)  abf/tmpbf bf16 during res stack; reused as yfwd f32 in DSS
  //   [128M,160M) ygelu bf16
  //   [160M,..)   packed bf16 weights + agg
  char* ws = (char*)d_ws;
  float* hbuf  = (float*)ws;
  float* yfwd  = (float*)(ws + (size_t)(64u << 20));
  bf16*  abf   = (bf16*)(ws + (size_t)(64u << 20));
  bf16*  tmpbf = (bf16*)(ws + (size_t)(96u << 20));
  bf16*  ygelu = (bf16*)(ws + (size_t)(128u << 20));
  bf16*  wres1 = (bf16*)(ws + (size_t)(160u << 20));
  bf16*  wres2 = wres1 + (long)NRES * HH * 768;
  bf16*  wglu  = wres2 + (long)NRES * HH * 768;
  float* agg   = (float*)(wglu + (long)NLAYERS * 2 * HH * HH);

  const int nres_el = NRES * HH * 768;
  const int nglu_el = NLAYERS * 2 * HH * HH;
  pack_conv3_kernel<<<(nres_el + 255) / 256, 256, 0, stream>>>(res_w1, wres1, nres_el);
  pack_conv3_kernel<<<(nres_el + 255) / 256, 256, 0, stream>>>(res_w2, wres2, nres_el);
  pack_glu_kernel<<<(nglu_el + 255) / 256, 256, 0, stream>>>(glu_w, wglu, nglu_el);

  enc_conv_kernel<<<dim3(LL / 64, BB), 256, 0, stream>>>(x, enc_w, enc_b, hbuf, abf);

  for (int r = 0; r < NRES; ++r) {
    // conv1: relu, bf16-only output
    conv3_wmma_kernel<false, false, true><<<dim3(LL / 32, BB), 256, 0, stream>>>(
        abf, wres1 + (long)r * HH * 768, res_b1 + r * HH,
        nullptr, nullptr, tmpbf);
    // conv2: + residual(hbuf), relu, writes f32 hbuf + bf16 abf
    conv3_wmma_kernel<true, true, true><<<dim3(LL / 32, BB), 256, 0, stream>>>(
        tmpbf, wres2 + (long)r * HH * 768, res_b2 + r * HH,
        hbuf, hbuf, abf);
  }

  for (int i = 0; i < NLAYERS; ++i) {
    dss_scan_kernel<<<(BB * HH) / 8, 256, 0, stream>>>(
        hbuf, yfwd, ygelu, log_dt + i * HH, lam_re + i * NN, lam_im + i * NN,
        W_re + (long)i * 2 * HH * NN, W_im + (long)i * 2 * HH * NN, Dp + i * HH);
    glu_ln_kernel<<<dim3(LL / 16, BB), 512, 0, stream>>>(
        ygelu, wglu + (long)i * 2 * HH * HH, glu_b + i * 2 * HH,
        ln_g + i * HH, ln_b + i * HH, hbuf);
  }

  float* out = (float*)d_out;
  float* dec_out = out;                           // (B, DOUT, L)
  float* seq_out = out + (long)BB * DOUT * LL;    // (B, H, L)
  float* reg_out = seq_out + (long)BB * HH * LL;  // (B, 1)

  species_agg_kernel<<<(BB * HH) / 8, 256, 0, stream>>>(hbuf, xs, sp_emb,
                                                        seq_out, agg);
  mlp_kernel<<<1, 256, 0, stream>>>(agg, h1w, h1b, h2w, h2b, h3w, h3b, h4w,
                                    h4b, reg_out);
  dec_conv_kernel<<<dim3(LL / 256, BB), 256, 0, stream>>>(hbuf, dec_w, dec_b,
                                                          dec_out);
}
